// LinearAttention3D_6734508720463
// MI455X (gfx1250) — matrix operand order repaired and checked by reading, not compiled, not run on hardware
//
#include <hip/hip_runtime.h>

// ---------------------------------------------------------------------------
// LinearAttention3D for MI455X (gfx1250, wave32, WMMA)
// Heavy math through v_wmma_f32_16x16x32_bf16 with f32 accumulation.
// ---------------------------------------------------------------------------

typedef __bf16 v16bf __attribute__((ext_vector_type(16)));
typedef __bf16 v4bf  __attribute__((ext_vector_type(4)));
typedef __bf16 v2bf  __attribute__((ext_vector_type(2)));
typedef float  v4f   __attribute__((ext_vector_type(4)));
typedef float  v8f   __attribute__((ext_vector_type(8)));

#define C_DIM 512
#define S_DIM 4096
#define NH    8
#define HD    64

// load 4 contiguous elements as f32x4 (vectorized: b128 for f32, b64 for bf16)
template <typename T>
__device__ inline v4f load4f(const T* p)
{
    if constexpr (sizeof(T) == 4) {
        return *(const v4f*)p;
    } else {
        v4bf b = *(const v4bf*)p;
        return (v4f){(float)b[0], (float)b[1], (float)b[2], (float)b[3]};
    }
}

// ---------------------------------------------------------------------------
// bf16 WMMA GEMM, block tile BM x 64, K-step 32, 256 threads (8 wave32).
//   BM=128: wave w -> m-subtile w, n-subtiles 0..3  (4 WMMA / wave / K-step)
//   BM=64 : wave w -> m-subtile w&3, n-subtiles {2*(w>>2), +1}
//   XTRANS: X is N-major (NxK row-major), i.e. B = X^T
//   YTRANS: store D transposed (Y[n*ldY + m])
//   KSPLIT: blockIdx.x selects a K-chunk; epilogue atomicAdd to f32 Y (n0=0)
// ---------------------------------------------------------------------------
template <int BM, typename WT, typename XT, typename OT,
          bool XTRANS, bool YTRANS, bool KSPLIT>
__global__ __launch_bounds__(256) void gemm_wmma_kernel(
    const WT* __restrict__ Wg, const XT* __restrict__ Xg,
    const float* __restrict__ bias, OT* __restrict__ Yg,
    int M, int N, int K, int kChunk, int ldW, int ldX, int ldY,
    long wStride, long xStride, long yStride)
{
    constexpr int NACC  = (BM == 128) ? 4 : 2;
    constexpr int WELEM = BM / 8;                 // W-tile elems per thread

    __shared__ __bf16 Ws[BM][36];                 // M-local x K-local (+pad)
    __shared__ __bf16 Xs[64][36];                 // N-local x K-local (+pad)

    const int tid  = threadIdx.x;
    const int lane = tid & 31;
    const int wave = tid >> 5;
    const int mi    = (BM == 128) ? wave : (wave & 3);
    const int nbase = (BM == 128) ? 0    : ((wave >> 2) * 2);

    const long z = blockIdx.z;
    const WT* Wp = Wg + z * wStride;
    const XT* Xp = Xg + z * xStride;
    OT*       Yp = Yg + z * yStride;

    const int m0 = blockIdx.y * BM;
    const int n0 = KSPLIT ? 0 : blockIdx.x * 64;
    const int kbeg = KSPLIT ? blockIdx.x * kChunk : 0;
    const int kend = KSPLIT ? min(K, kbeg + kChunk) : K;

    // K-major X staging mapping: 4 cols from row k and row k+1 per thread
    const int xn4 = (tid & 15) * 4;               // 0..60
    const int xk  = (tid >> 4) * 2;               // 0..30 (even)

    v8f acc[NACC];
    #pragma unroll
    for (int t = 0; t < NACC; ++t)
        acc[t] = (v8f){0.f, 0.f, 0.f, 0.f, 0.f, 0.f, 0.f, 0.f};

    for (int k0 = kbeg; k0 < kend; k0 += 32) {
        // ---- stage W tile: BM rows (m) x 32 cols (k) ----
        #pragma unroll
        for (int i = 0; i < WELEM; ++i) {
            int idx = tid * WELEM + i;            // 0 .. BM*32-1
            int r = idx >> 5, c = idx & 31;
            Ws[r][c] = (__bf16)(float)Wp[(long)(m0 + r) * ldW + (k0 + c)];
        }
        // ---- stage X tile into Xs[n][k] ----
        if (XTRANS) {
            // X is N-major: contiguous 8 halves along k per thread
            #pragma unroll
            for (int i = 0; i < 8; ++i) {
                int idx = tid * 8 + i;
                int r = idx >> 5, c = idx & 31;   // r=n-local, c=k-local
                Xs[r][c] = (__bf16)(float)Xp[(long)(n0 + r) * ldX + (k0 + c)];
            }
        } else {
            // X is K-major: two vector loads (rows k, k+1; 4 cols each),
            // pack four bf16 (k,k+1) pairs -> four ds_store_b32.
            const XT* b0 = &Xp[(long)(k0 + xk) * ldX + (n0 + xn4)];
            v4f r0 = load4f(b0);
            v4f r1 = load4f(b0 + ldX);
            #pragma unroll
            for (int j = 0; j < 4; ++j)
                *(v2bf*)&Xs[xn4 + j][xk] = (v2bf){(__bf16)r0[j], (__bf16)r1[j]};
        }
        // prefetch next K-slice (global_prefetch_b8; speculative, OOB ok)
        if (k0 + 32 < kend) {
            __builtin_prefetch(&Wp[(long)(m0 + ((tid * WELEM) >> 5)) * ldW + (k0 + 32)], 0, 3);
            if (XTRANS)
                __builtin_prefetch(&Xp[(long)(n0 + ((tid * 8) >> 5)) * ldX + (k0 + 32)], 0, 3);
            else
                __builtin_prefetch(&Xp[(long)(k0 + 32 + xk) * ldX + (n0 + xn4)], 0, 3);
        }
        __syncthreads();

        // ---- A fragment (16x32 bf16) per ISA wave32 layout ----
        v16bf af;
        const int rA = mi * 16 + (lane & 15);
        #pragma unroll
        for (int h = 0; h < 16; ++h) {
            int kk = (h < 8) ? ((lane < 16 ? 0 : 8) + h)
                             : ((lane < 16 ? 16 : 24) + (h - 8));
            af[h] = Ws[rA][kk];
        }
        // ---- B fragments (32x16): lanes 0-15 K=0..7,16..23, lanes 16-31 K=8..15,24..31 ----
        v16bf bf[NACC];
        #pragma unroll
        for (int t = 0; t < NACC; ++t) {
            const int rB = (nbase + t) * 16 + (lane & 15);
            #pragma unroll
            for (int h = 0; h < 16; ++h) {
                int kk = (h < 8) ? ((lane < 16 ? 0 : 8) + h) : ((lane < 16 ? 16 : 24) + (h - 8));
                bf[t][h] = Xs[rB][kk];
            }
        }
        #pragma unroll
        for (int t = 0; t < NACC; ++t)
            acc[t] = __builtin_amdgcn_wmma_f32_16x16x32_bf16(
                false, af, false, bf[t], (short)0, acc[t], false, false);
        __syncthreads();
    }

    // ---- epilogue: D layout = 8 VGPRs, lanes 0-15 M=r, lanes 16-31 M=8+r ----
    const int colL = lane & 15;
    #pragma unroll
    for (int r = 0; r < 8; ++r) {
        int rowL = (lane < 16) ? r : (8 + r);
        int gm = m0 + mi * 16 + rowL;
        float bv = (!KSPLIT && bias) ? bias[gm] : 0.0f;
        #pragma unroll
        for (int t = 0; t < NACC; ++t) {
            int gn = n0 + (nbase + t) * 16 + colL;
            float v = acc[t][r] + bv;
            if constexpr (KSPLIT) {
                atomicAdd((float*)&Yp[(long)gm * ldY + gn], v);
            } else if constexpr (YTRANS) {
                Yp[(long)gn * ldY + gm] = (OT)v;
            } else {
                Yp[(long)gm * ldY + gn] = (OT)v;
            }
        }
    }
}

// ---------------------------------------------------------------------------
// Softmax over the feature (head_dim) axis, in-place on bf16 q; folds in the
// d^-0.5 = 0.125 scale.  One thread per (b, head, s) column of 64 elements.
// ---------------------------------------------------------------------------
__global__ __launch_bounds__(256) void softmax_feature_kernel(__bf16* __restrict__ q)
{
    long tid = (long)blockIdx.x * blockDim.x + threadIdx.x;   // 0 .. 64*4096-1
    long zz = tid >> 12;        // (b*NH + n)
    long s  = tid & 4095;
    __bf16* p = q + zz * (long)(HD * S_DIM) + s;

    float v[HD];
    float mx = -1e30f;
    #pragma unroll
    for (int i = 0; i < HD; ++i) {
        v[i] = (float)p[(long)i * S_DIM];
        mx = fmaxf(mx, v[i]);
    }
    float sum = 0.f;
    #pragma unroll
    for (int i = 0; i < HD; ++i) {
        v[i] = __expf(v[i] - mx);
        sum += v[i];
    }
    float inv = 0.125f / sum;
    #pragma unroll
    for (int i = 0; i < HD; ++i)
        p[(long)i * S_DIM] = (__bf16)(v[i] * inv);
}

// ---------------------------------------------------------------------------
// Softmax over the sequence axis (4096), in-place on bf16 k.
// One 256-thread block per row; LDS tree reductions for max and sum.
// ---------------------------------------------------------------------------
__global__ __launch_bounds__(256) void softmax_seq_kernel(__bf16* __restrict__ k)
{
    __shared__ float red[256];
    __bf16* p = k + (long)blockIdx.x * S_DIM;
    const int t = threadIdx.x;

    float v[16];
    float mx = -1e30f;
    #pragma unroll
    for (int i = 0; i < 16; ++i) {
        v[i] = (float)p[t + 256 * i];
        mx = fmaxf(mx, v[i]);
    }
    red[t] = mx;
    __syncthreads();
    for (int off = 128; off > 0; off >>= 1) {
        if (t < off) red[t] = fmaxf(red[t], red[t + off]);
        __syncthreads();
    }
    mx = red[0];
    __syncthreads();

    float sum = 0.f;
    #pragma unroll
    for (int i = 0; i < 16; ++i) {
        v[i] = __expf(v[i] - mx);
        sum += v[i];
    }
    red[t] = sum;
    __syncthreads();
    for (int off = 128; off > 0; off >>= 1) {
        if (t < off) red[t] += red[t + off];
        __syncthreads();
    }
    float inv = 1.0f / red[0];
    #pragma unroll
    for (int i = 0; i < 16; ++i)
        p[t + 256 * i] = (__bf16)(v[i] * inv);
}

// ---------------------------------------------------------------------------
// Zero-init for the f32 context accumulator (graph-capture safe).
// ---------------------------------------------------------------------------
__global__ __launch_bounds__(256) void zero_f32_kernel(float* __restrict__ p, long n)
{
    long i = (long)blockIdx.x * blockDim.x + threadIdx.x;
    if (i < n) p[i] = 0.f;
}

// ---------------------------------------------------------------------------
// ctx convert: f32 (z, i, j) -> bf16 transposed (z, j, i)
// ---------------------------------------------------------------------------
__global__ __launch_bounds__(256) void ctx_cvt_kernel(const float* __restrict__ cf,
                                                      __bf16* __restrict__ cb)
{
    long t = (long)blockIdx.x * blockDim.x + threadIdx.x;   // 0 .. 64*64*64-1
    long z = t >> 12;
    long i = (t >> 6) & 63;
    long j = t & 63;
    cb[(z << 12) + (j << 6) + i] = (__bf16)cf[t];
}

// ---------------------------------------------------------------------------
// Host-side orchestration (graph-capture safe: kernel launches only).
// ---------------------------------------------------------------------------
extern "C" void kernel_launch(void* const* d_in, const int* in_sizes, int n_in,
                              void* d_out, int out_size, void* d_ws, size_t ws_size,
                              hipStream_t stream)
{
    (void)in_sizes; (void)n_in; (void)out_size; (void)ws_size;
    const float* x  = (const float*)d_in[0];
    const float* wq = (const float*)d_in[1];
    const float* bq = (const float*)d_in[2];
    const float* wk = (const float*)d_in[3];
    const float* bk = (const float*)d_in[4];
    const float* wv = (const float*)d_in[5];
    const float* bv = (const float*)d_in[6];
    const float* wo = (const float*)d_in[7];
    const float* bo = (const float*)d_in[8];
    float* out = (float*)d_out;

    const long BCS = 8L * C_DIM * S_DIM;     // 16,777,216 elements
    char* ws = (char*)d_ws;
    __bf16* qb    = (__bf16*)(ws);                        // 2*BCS bytes
    __bf16* kb    = (__bf16*)(ws + 2 * BCS);
    __bf16* vb    = (__bf16*)(ws + 4 * BCS);
    __bf16* attn  = (__bf16*)(ws + 6 * BCS);
    __bf16* ctxb  = (__bf16*)(ws + 8 * BCS);              // 512 KB
    float*  ctxf  = (float*)(ws + 8 * BCS + (1L << 20));  // 1 MB

    const long cs = (long)C_DIM * S_DIM;                  // per-batch plane
    const long hs = (long)HD * S_DIM;                     // per-(b,head) plane
    const long CTX_N = 64L * HD * HD;                     // 262144

    dim3 blk(256);
    dim3 gproj(S_DIM / 64, C_DIM / 128, 8);               // (64, 4, 8), z = batch

    // q/k/v projections: 128x64 tiles, f32 in, bf16 out, +bias
    gemm_wmma_kernel<128, float, float, __bf16, false, false, false>
        <<<gproj, blk, 0, stream>>>(wq, x, bq, qb, C_DIM, S_DIM, C_DIM, C_DIM,
                                    C_DIM, S_DIM, S_DIM, 0L, cs, cs);
    gemm_wmma_kernel<128, float, float, __bf16, false, false, false>
        <<<gproj, blk, 0, stream>>>(wk, x, bk, kb, C_DIM, S_DIM, C_DIM, C_DIM,
                                    C_DIM, S_DIM, S_DIM, 0L, cs, cs);
    gemm_wmma_kernel<128, float, float, __bf16, false, false, false>
        <<<gproj, blk, 0, stream>>>(wv, x, bv, vb, C_DIM, S_DIM, C_DIM, C_DIM,
                                    C_DIM, S_DIM, S_DIM, 0L, cs, cs);

    // softmax(q, axis=d) * d^-0.5   and   softmax(k, axis=s), both in place
    softmax_feature_kernel<<<(64 * S_DIM) / 256, blk, 0, stream>>>(qb);
    softmax_seq_kernel<<<8 * C_DIM, blk, 0, stream>>>(kb);

    // context[b,n] = k (64 x 4096) * v^T (4096 x 64), K split 16 ways,
    // f32 atomic accumulation, then convert to bf16 transposed (j,i).
    zero_f32_kernel<<<(CTX_N + 255) / 256, blk, 0, stream>>>(ctxf, CTX_N);
    gemm_wmma_kernel<64, __bf16, __bf16, float, true, false, true>
        <<<dim3(16, 1, 64), blk, 0, stream>>>(kb, vb, nullptr, ctxf,
                                              HD, HD, S_DIM, S_DIM / 16,
                                              S_DIM, S_DIM, HD, hs, hs,
                                              (long)(HD * HD));
    ctx_cvt_kernel<<<(CTX_N + 255) / 256, blk, 0, stream>>>(ctxf, ctxb);

    // out[b,n] = ctx^T (64 x 64) * q (64 x 4096)
    gemm_wmma_kernel<64, __bf16, __bf16, __bf16, false, false, false>
        <<<dim3(S_DIM / 64, 1, 64), blk, 0, stream>>>(ctxb, qb, nullptr, attn,
                                                      HD, S_DIM, HD, HD,
                                                      HD, S_DIM, S_DIM,
                                                      (long)(HD * HD), hs, hs);

    // output projection: wo (512x512) * attn + bo -> f32 d_out
    gemm_wmma_kernel<128, float, __bf16, float, false, false, false>
        <<<gproj, blk, 0, stream>>>(wo, attn, bo, out, C_DIM, S_DIM, C_DIM,
                                    C_DIM, C_DIM, S_DIM, S_DIM, 0L, cs, cs);
}
